// Model_52518860096554
// MI455X (gfx1250) — matrix operand order repaired and checked by reading, not compiled, not run on hardware
//
#include <hip/hip_runtime.h>

// ---------------------------------------------------------------------------
// Problem constants (B=1, H=32, S=2048, E=512, head_dim = 32)
// ---------------------------------------------------------------------------
#define HH 32
#define SS 2048
#define EE 512
#define DD 32
#define NEGF (-1.442695041e9f)   // -1e9 * log2(e), mask fill in log2 domain

typedef __attribute__((ext_vector_type(16))) __bf16 v16bf;
typedef __attribute__((ext_vector_type(8)))  __bf16 v8bf;
typedef __attribute__((ext_vector_type(8)))  float  v8f;
typedef __attribute__((ext_vector_type(4)))  float  v4f;

// A-operand (16-bit, 16x32): per-lane 16 contiguous K values
// (caller passes p already offset by hi*16 so high lanes get K 16-31)
__device__ __forceinline__ v16bf cvtA16(const float* p) {
    v4f a0 = *(const v4f*)(p + 0);
    v4f a1 = *(const v4f*)(p + 4);
    v4f a2 = *(const v4f*)(p + 8);
    v4f a3 = *(const v4f*)(p + 12);
    v16bf r;
#pragma unroll
    for (int i = 0; i < 4; ++i) {
        r[i]      = (__bf16)a0[i];
        r[4 + i]  = (__bf16)a1[i];
        r[8 + i]  = (__bf16)a2[i];
        r[12 + i] = (__bf16)a3[i];
    }
    return r;
}

// B-operand (16-bit, 32x16): per-lane 16 contiguous K values (low lanes K0-15,
// high lanes K16-31; caller passes p offset by hi*16), from f32 source.
__device__ __forceinline__ v16bf cvtB16(const float* p) {
    v4f a0 = *(const v4f*)(p + 0);
    v4f a1 = *(const v4f*)(p + 4);
    v4f a2 = *(const v4f*)(p + 8);
    v4f a3 = *(const v4f*)(p + 12);
    v16bf r;
#pragma unroll
    for (int i = 0; i < 4; ++i) {
        r[i]      = (__bf16)a0[i];
        r[4 + i]  = (__bf16)a1[i];
        r[8 + i]  = (__bf16)a2[i];
        r[12 + i] = (__bf16)a3[i];
    }
    return r;
}

// bf16 A-operand load: 16 contiguous elements from p
__device__ __forceinline__ v16bf ldA_bf(const __bf16* p) {
    v8bf lo = *(const v8bf*)(p);
    v8bf hi = *(const v8bf*)(p + 8);
    return __builtin_shufflevector(lo, hi, 0,1,2,3,4,5,6,7,8,9,10,11,12,13,14,15);
}

// bf16 B-operand load: 16 contiguous elements from p
__device__ __forceinline__ v16bf ldB_bf(const __bf16* p) {
    v8bf lo = *(const v8bf*)(p);
    v8bf hi = *(const v8bf*)(p + 8);
    return __builtin_shufflevector(lo, hi, 0,1,2,3,4,5,6,7,8,9,10,11,12,13,14,15);
}

// ---------------------------------------------------------------------------
// Projection: Out[h,s,0:32] = X[h,s,0:512] @ W^T + bias, f32 -> bf16.
// grid = (S/128, H), block = 256 (8 waves). Wave -> 16 rows x 32 cols.
// transpose_out==1 stores Out as [h][d][s] (for V, feeding WMMA B-operand).
// HBM-bound: streams X exactly once.
// ---------------------------------------------------------------------------
__global__ __launch_bounds__(256) void proj_kernel(
    const float* __restrict__ X, const float* __restrict__ W,
    const float* __restrict__ bias, __bf16* __restrict__ Out,
    int transpose_out)
{
    const int wid  = threadIdx.x >> 5;
    const int lane = threadIdx.x & 31;
    const int lm   = lane & 15;
    const int hi   = lane >> 4;
    const int head = blockIdx.y;
    const int rbase = blockIdx.x * 128 + wid * 16;

    const float* Xh   = X + (size_t)head * SS * EE;
    const float* arow = Xh + (size_t)(rbase + lm) * EE;   // A: row = lane%16
    const float* w0   = W + (size_t)lm * EE;              // B tile 0: n = lm
    const float* w1   = W + (size_t)(lm + 16) * EE;       // B tile 1: n = lm+16

    v8f acc0 = {}; v8f acc1 = {};
#pragma unroll 2
    for (int e = 0; e < EE; e += 32) {
        v16bf a  = cvtA16(arow + e + hi * 16);
        v16bf b0 = cvtB16(w0 + e + hi * 16);
        v16bf b1 = cvtB16(w1 + e + hi * 16);
        acc0 = __builtin_amdgcn_wmma_f32_16x16x32_bf16(false, a, false, b0,
                                                       (short)0, acc0, false, false);
        acc1 = __builtin_amdgcn_wmma_f32_16x16x32_bf16(false, a, false, b1,
                                                       (short)0, acc1, false, false);
    }

    const float bv0 = bias[lm];
    const float bv1 = bias[lm + 16];
    if (!transpose_out) {
        // row-major [h][s][32]; C layout: row = r + 8*hi, col = lm / lm+16
        __bf16* oh = Out + (size_t)head * SS * DD;
#pragma unroll
        for (int r = 0; r < 8; ++r) {
            const int row = rbase + r + hi * 8;
            __bf16* orow = oh + (size_t)row * DD;
            orow[lm]      = (__bf16)(acc0[r] + bv0);
            orow[lm + 16] = (__bf16)(acc1[r] + bv1);
        }
    } else {
        // transposed [h][d][s]
        __bf16* oh = Out + (size_t)head * DD * SS;
#pragma unroll
        for (int r = 0; r < 8; ++r) {
            const int row = rbase + r + hi * 8;
            oh[(size_t)lm * SS + row]        = (__bf16)(acc0[r] + bv0);
            oh[(size_t)(lm + 16) * SS + row] = (__bf16)(acc1[r] + bv1);
        }
    }
}

// ---------------------------------------------------------------------------
// Flash attention over projected bf16 tensors.
// grid = (S/128, H), block = 256 (8 waves). Wave -> 16 query rows.
// K/V tiles for each 32-key chunk are staged once per workgroup into
// double-buffered LDS (256 threads x one b128 each) so the 8 waves share
// them via ds_load_b128 instead of 8x redundant L2 traffic.
// One s_barrier per chunk: stores into buf (cur^1) are safe after the
// previous barrier; the end-of-iteration barrier publishes them.
// ---------------------------------------------------------------------------
__global__ __launch_bounds__(256) void attn_kernel(
    const __bf16* __restrict__ Qp,   // [h][s][32]
    const __bf16* __restrict__ Kp,   // [h][s][32]
    const __bf16* __restrict__ Vt,   // [h][32][s]
    const unsigned char* __restrict__ mask,   // [s][s]
    const float* __restrict__ inv_scale,      // [h]
    float* __restrict__ out)                   // [h][s][32]
{
    __shared__ __align__(16) __bf16 Kbuf[2][32][40];   // keys x d, 80B rows
    __shared__ __align__(16) __bf16 Vbuf[2][32][40];   // d x keys, 80B rows
    __shared__ __align__(16) __bf16 Pbuf[8][16][40];   // per-wave P transpose

    const int tid  = threadIdx.x;
    const int wid  = tid >> 5;
    const int lane = tid & 31;
    const int lm   = lane & 15;
    const int hi   = lane >> 4;
    const int head = blockIdx.y;
    const int rbase = blockIdx.x * 128 + wid * 16;

    const float sc = 1.442695041f / inv_scale[head];   // fold log2(e)

    const __bf16* qh = Qp + (size_t)head * SS * DD;
    const __bf16* kh = Kp + (size_t)head * SS * DD;
    const __bf16* vh = Vt + (size_t)head * DD * SS;

    // ---- staging role: threads 0-127 fill K tile, 128-255 fill V tile ----
    const int  sj    = tid & 127;
    const int  srow  = sj >> 2;          // 0..31 (key row for K, d row for V)
    const int  spart = (sj & 3) * 8;     // half-word offset 0/8/16/24
    const bool isV   = tid >= 128;
    const __bf16* sbase = isV ? (vh + (size_t)srow * SS + spart)
                              : (kh + (size_t)srow * DD + spart);
    const int schunk = isV ? 32 : 32 * DD;   // element advance per 32-key chunk

    // Q A-tile: row = lm, K(d) split by lane half
    const v16bf qa = ldA_bf(qh + (size_t)(rbase + lm) * DD + hi * 16);

    float m[8], l[8], alpha[8];
    v8f o0 = {}; v8f o1 = {};
#pragma unroll
    for (int r = 0; r < 8; ++r) { m[r] = -3.0e38f; l[r] = 0.0f; }

    // preload chunk 0
    {
        v8bf stg = *(const v8bf*)(sbase);
        *(v8bf*)(isV ? &Vbuf[0][srow][spart] : &Kbuf[0][srow][spart]) = stg;
    }
    __syncthreads();

    const int NCH = SS / 32;
#pragma unroll 1
    for (int i = 0; i < NCH; ++i) {
        const int t   = i * 32;
        const int cur = i & 1;
        const bool havenext = (i + 1 < NCH);

        // issue global load for next chunk early (latency hidden by compute)
        v8bf stg = {};
        if (havenext) stg = *(const v8bf*)(sbase + (size_t)(i + 1) * schunk);

        // K^T B-tiles from LDS: N = key lm (+16), K(d) = hi*16..hi*16+15
        v16bf kb0 = ldB_bf(&Kbuf[cur][lm][hi * 16]);
        v16bf kb1 = ldB_bf(&Kbuf[cur][16 + lm][hi * 16]);
        v8f z = {};
        v8f s0 = __builtin_amdgcn_wmma_f32_16x16x32_bf16(false, qa, false, kb0,
                                                         (short)0, z, false, false);
        v8f s1 = __builtin_amdgcn_wmma_f32_16x16x32_bf16(false, qa, false, kb1,
                                                         (short)0, z, false, false);

        const unsigned char* mbase = mask + (size_t)(rbase + hi * 8) * SS + t;
#pragma unroll
        for (int r = 0; r < 8; ++r) {
            float a = s0[r] * sc;
            float b = s1[r] * sc;
            const unsigned char* mr = mbase + (size_t)r * SS;
            if (mr[lm])      a = NEGF;
            if (mr[16 + lm]) b = NEGF;
            float mx = fmaxf(a, b);
            mx = fmaxf(mx, __shfl_xor(mx, 1, 32));
            mx = fmaxf(mx, __shfl_xor(mx, 2, 32));
            mx = fmaxf(mx, __shfl_xor(mx, 4, 32));
            mx = fmaxf(mx, __shfl_xor(mx, 8, 32));
            const float mnew = fmaxf(m[r], mx);
            const float al = exp2f(m[r] - mnew);
            const float p0 = exp2f(a - mnew);
            const float p1 = exp2f(b - mnew);
            float rs = p0 + p1;
            rs += __shfl_xor(rs, 1, 32);
            rs += __shfl_xor(rs, 2, 32);
            rs += __shfl_xor(rs, 4, 32);
            rs += __shfl_xor(rs, 8, 32);
            m[r] = mnew;
            l[r] = l[r] * al + rs;
            alpha[r] = al;
            const int row = r + hi * 8;          // C-layout row of this lane
            Pbuf[wid][row][lm]      = (__bf16)p0;
            Pbuf[wid][row][16 + lm] = (__bf16)p1;
        }
        asm volatile("s_wait_dscnt 0" ::: "memory");

        // P A-tile from LDS (row = lm, K split by lane half)
        v16bf pa = ldA_bf(&Pbuf[wid][lm][0] + hi * 16);

#pragma unroll
        for (int r = 0; r < 8; ++r) { o0[r] *= alpha[r]; o1[r] *= alpha[r]; }

        // V B-tiles from LDS: N = d lm (+16), K(keys) = hi*16..hi*16+15
        v16bf vb0 = ldB_bf(&Vbuf[cur][lm][hi * 16]);
        v16bf vb1 = ldB_bf(&Vbuf[cur][16 + lm][hi * 16]);
        o0 = __builtin_amdgcn_wmma_f32_16x16x32_bf16(false, pa, false, vb0,
                                                     (short)0, o0, false, false);
        o1 = __builtin_amdgcn_wmma_f32_16x16x32_bf16(false, pa, false, vb1,
                                                     (short)0, o1, false, false);

        // publish next chunk into the other buffer (readers of cur^1 finished
        // before the previous barrier)
        if (havenext) {
            const int nb = cur ^ 1;
            *(v8bf*)(isV ? &Vbuf[nb][srow][spart] : &Kbuf[nb][srow][spart]) = stg;
        }
        __syncthreads();
    }

    float* ohead = out + (size_t)head * SS * DD;
#pragma unroll
    for (int r = 0; r < 8; ++r) {
        const float inv = 1.0f / l[r];
        const int row = rbase + r + hi * 8;
        float* orow = ohead + (size_t)row * DD;
        orow[lm]      = o0[r] * inv;
        orow[lm + 16] = o1[r] * inv;
    }
}

// ---------------------------------------------------------------------------
// Launch
// ---------------------------------------------------------------------------
extern "C" void kernel_launch(void* const* d_in, const int* in_sizes, int n_in,
                              void* d_out, int out_size, void* d_ws, size_t ws_size,
                              hipStream_t stream) {
    const float* query = (const float*)d_in[0];
    const float* key   = (const float*)d_in[1];
    const float* value = (const float*)d_in[2];
    const unsigned char* mask = (const unsigned char*)d_in[3];
    const float* Wq = (const float*)d_in[4];
    const float* bq = (const float*)d_in[5];
    const float* Wk = (const float*)d_in[6];
    const float* bk = (const float*)d_in[7];
    const float* Wv = (const float*)d_in[8];
    const float* bv = (const float*)d_in[9];
    const float* inv_scale = (const float*)d_in[10];

    const size_t perTensor = (size_t)HH * SS * DD;     // bf16 elements
    __bf16* Qp = (__bf16*)d_ws;
    __bf16* Kp = Qp + perTensor;
    __bf16* Vt = Kp + perTensor;

    dim3 blk(256);
    dim3 grid(SS / 128, HH);
    proj_kernel<<<grid, blk, 0, stream>>>(query, Wq, bq, Qp, 0);
    proj_kernel<<<grid, blk, 0, stream>>>(key,   Wk, bk, Kp, 0);
    proj_kernel<<<grid, blk, 0, stream>>>(value, Wv, bv, Vt, 1);
    attn_kernel<<<grid, blk, 0, stream>>>(Qp, Kp, Vt, mask, inv_scale, (float*)d_out);
}
